// Pyramid_47528108098249
// MI455X (gfx1250) — compile-verified, matrix operand order checked by reading, not run on hardware
//
#include <hip/hip_runtime.h>
#include <hip/hip_bf16.h>

// CDNA5 wave32 WMMA types
typedef __attribute__((ext_vector_type(16))) __bf16 v16bf;
typedef __attribute__((ext_vector_type(8)))  float  v8f;

#define SEQ    64
#define MODEL  256
#define HALFD  128
#define GATES  640   // 5 * HALFD
#define WORDD  300
#define MLPD   1024
#define NCLS   3

// f32 -> bf16 via native fptrunc (backend emits hardware cvt, RNE)
__device__ __forceinline__ __bf16 f2bf(float f) { return (__bf16)f; }

__device__ __forceinline__ float sigm(float x) { return 1.f / (1.f + __expf(-x)); }
__device__ __forceinline__ float tanh_fast(float x) {
  float e = __expf(-2.f * x);
  return (1.f - e) / (1.f + e);
}

// ---------------------------------------------------------------------------
// Prep: W_comp (256 x 640, f32 row-major) -> bf16 B-fragments in the CDNA5
// 16x16x32 bf16 B lane layout.  Fragment index: [n_tile 0..39][k_tile 0..7]
// [lane 0..31][elem 0..15], where lane holds column n = n_tile*16 + lane%16
// and elem v holds K = k_tile*32 + v + 8*(lane/16) + (v>=8 ? 8 : 0).
// ---------------------------------------------------------------------------
__global__ void prep_wcomp_kernel(const float* __restrict__ W_comp,
                                  __bf16* __restrict__ Bfrag) {
  int bt   = blockIdx.x;          // 0..319 == nt*8 + kt
  int nt   = bt >> 3;
  int kt   = bt & 7;
  int tid  = threadIdx.x;         // 0..511
  int lane = tid >> 4;            // 0..31
  int v    = tid & 15;            // 0..15
  int half = lane >> 4;
  int kp   = v + 8 * half + ((v >= 8) ? 8 : 0);
  int k    = kt * 32 + kp;                       // 0..255
  int n    = nt * 16 + (lane & 15);              // 0..639
  Bfrag[(size_t)(((nt * 8 + kt) * 32 + lane) * 16 + v)] = f2bf(W_comp[k * GATES + n]);
}

// ---------------------------------------------------------------------------
// Fused pyramid kernel: one workgroup per batch element, 8 waves.
// LDS: S (65x256 f32 states, row 64 = zero pad), C (64x256 comp h|c),
//      softmax scratch, MLP activations.  ~141 KB dynamic LDS; one WG/WGP.
// ---------------------------------------------------------------------------
__global__ void __launch_bounds__(256)
pyramid_kernel(const int*   __restrict__ sentences,
               const float* __restrict__ embed,
               const float* __restrict__ W_enc,  const float* __restrict__ b_enc,
               const float* __restrict__ b_comp,
               const float* __restrict__ W_sel,  const float* __restrict__ b_sel,
               const float* __restrict__ W1,     const float* __restrict__ b1,
               const float* __restrict__ W2,     const float* __restrict__ b2,
               const float* __restrict__ W_out,  const float* __restrict__ b_out,
               const __bf16* __restrict__ Bfrag,
               float* __restrict__ out) {
  extern __shared__ float sm[];
  float* S    = sm;                       // 65*256 states (h|c per position)
  float* C    = S + 65 * MODEL;           // 64*256 comp   (h|c per pair)
  float* lgts = C + 64 * MODEL;           // 64
  float* prb  = lgts + 64;                // 64
  float* cpl  = prb + 64;                 // 64
  float* cpr  = cpl + 64;                 // 64
  float* h1   = cpr + 64;                 // 1024
  float* h2   = h1 + MLPD;                // 1024

  const int b    = blockIdx.x;
  const int tid  = threadIdx.x;
  const int lane = tid & 31;
  const int wv   = tid >> 5;              // wave 0..7
  const int half = lane >> 4;
  const int lr   = lane & 15;

  // ---------------- encoder: S[p][d] = emb[sent[b][p]] @ W_enc + b_enc ------
  {
    const int d = tid;                    // column 0..255
    for (int p = 0; p < SEQ; ++p) {
      int tok = sentences[b * SEQ + p];
      const float* er = embed + (size_t)tok * WORDD;
      float acc = b_enc[d];
      for (int k = 0; k < WORDD; ++k)
        acc = fmaf(er[k], W_enc[k * MODEL + d], acc);
      S[p * MODEL + d] = acc;
    }
    S[SEQ * MODEL + tid] = 0.f;           // zero pad row 64
  }
  __syncthreads();

  // ---------------- 63 sequential tree-LSTM layers --------------------------
  for (int n = SEQ - 1; n >= 1; --n) {
    const int mt = wv >> 1;               // m-tile owned by this wave
    if (mt * 16 < n) {                    // wave-uniform guard: EXEC all-1 inside
      // Build bf16 A fragments (A = [h_l | h_r], K = 256) once per m-tile.
      const int row = mt * 16 + lr;
      v16bf afr[8];
#pragma unroll
      for (int kt = 0; kt < 8; ++kt) {
#pragma unroll
        for (int v = 0; v < 16; ++v) {
          int kp = v + 8 * half + ((v >= 8) ? 8 : 0);
          int k  = kt * 32 + kp;
          float val = (k < HALFD) ? S[row * MODEL + k]
                                  : S[(row + 1) * MODEL + (k - HALFD)];
          afr[kt][v] = f2bf(val);
        }
      }
      // Each wave covers 4 j-tiles (hidden-unit tiles), all 5 gates fused.
#pragma unroll
      for (int jj = 0; jj < 4; ++jj) {
        const int jt = (wv & 1) + 2 * jj;        // 0..7
        const int j  = jt * 16 + lr;             // hidden unit 0..127
        // per-gate biases: splat into WMMA C accumulator (all 8 rows share col j)
        float bias[5];
#pragma unroll
        for (int g = 0; g < 5; ++g) bias[g] = b_comp[g * HALFD + j];
        v8f acc[5];
#pragma unroll
        for (int g = 0; g < 5; ++g) {
          const float bg = bias[g];
          v8f a = {bg, bg, bg, bg, bg, bg, bg, bg};
          const int nt = g * 8 + jt;             // column tile in 640-wide gates
          const __bf16* bp = Bfrag + (size_t)(nt * 8) * 512 + lane * 16;
#pragma unroll
          for (int kt = 0; kt < 8; ++kt) {
            v16bf bfr = *(const v16bf*)(bp + kt * 512);
            a = __builtin_amdgcn_wmma_f32_16x16x32_bf16(
                    false, afr[kt], false, bfr, (short)0, a, false, false);
          }
          acc[g] = a;
        }
        // D layout: VGPR r -> row m = r + 8*half, lane%16 -> column j.
#pragma unroll
        for (int r = 0; r < 8; ++r) {
          const int p = mt * 16 + r + 8 * half;
          float cll = S[p * MODEL + HALFD + j];
          float crr = S[(p + 1) * MODEL + HALFD + j];
          float av  = acc[0][r];
          float iv  = acc[1][r];
          float f1v = acc[2][r];
          float f2v = acc[3][r];
          float ov  = acc[4][r];
          float cc  = tanh_fast(av) * sigm(iv) + sigm(f1v) * cll + sigm(f2v) * crr;
          float hh  = sigm(ov) * tanh_fast(cc);
          C[p * MODEL + j]         = hh;
          C[p * MODEL + HALFD + j] = cc;
        }
      }
    }
    __syncthreads();

    // selection logits: one thread per pair position
    if (tid < n) {
      float acc = b_sel[0];
      for (int k = 0; k < MODEL; ++k)
        acc = fmaf(C[tid * MODEL + k], W_sel[k], acc);
      lgts[tid] = acc;
    }
    __syncthreads();

    // softmax + cumsum over <=63 positions: trivial serial pass
    if (tid == 0) {
      float mx = -3.4e38f;
      for (int p = 0; p < n; ++p) mx = fmaxf(mx, lgts[p]);
      float s = 0.f;
      for (int p = 0; p < n; ++p) { float e = __expf(lgts[p] - mx); prb[p] = e; s += e; }
      float inv = 1.f / s, cs = 0.f;
      for (int p = 0; p < n; ++p) {
        float q = prb[p] * inv;
        prb[p] = q;
        cpr[p] = cs;          // copy_right = prefix-exclusive cumsum
        cs += q;
        cpl[p] = 1.f - cs;    // copy_left  = 1 - inclusive cumsum
      }
    }
    __syncthreads();

    // in-place convex state update; each thread owns one column d
    {
      const int d = tid;
      float oldp = S[d];
      for (int p = 0; p < n; ++p) {
        float oldp1 = S[(p + 1) * MODEL + d];
        S[p * MODEL + d] = cpl[p] * oldp + cpr[p] * oldp1 + prb[p] * C[p * MODEL + d];
        oldp = oldp1;
      }
    }
    __syncthreads();
  }

  // ---------------- MLP head on final state (S row 0) ------------------------
  {
#pragma unroll
    for (int rr = 0; rr < 4; ++rr) {
      int jj = tid + 256 * rr;
      float acc = b1[jj];
      for (int k = 0; k < MODEL; ++k)
        acc = fmaf(S[k], W1[k * MLPD + jj], acc);
      h1[jj] = fmaxf(acc, 0.f);
    }
    __syncthreads();
#pragma unroll
    for (int rr = 0; rr < 4; ++rr) {
      int jj = tid + 256 * rr;
      float acc = b2[jj];
      for (int k = 0; k < MLPD; ++k)
        acc = fmaf(h1[k], W2[k * MLPD + jj], acc);
      h2[jj] = fmaxf(acc, 0.f);
    }
    __syncthreads();
    if (tid < NCLS) {
      float acc = b_out[tid];
      for (int k = 0; k < MLPD; ++k)
        acc = fmaf(h2[k], W_out[k * NCLS + tid], acc);
      out[b * NCLS + tid] = acc;
    }
  }
}

extern "C" void kernel_launch(void* const* d_in, const int* in_sizes, int n_in,
                              void* d_out, int out_size, void* d_ws, size_t ws_size,
                              hipStream_t stream) {
  const int*   sentences = (const int*)  d_in[0];
  const float* embed     = (const float*)d_in[1];
  const float* W_enc     = (const float*)d_in[2];
  const float* b_enc     = (const float*)d_in[3];
  const float* W_comp    = (const float*)d_in[4];
  const float* b_comp    = (const float*)d_in[5];
  const float* W_sel     = (const float*)d_in[6];
  const float* b_sel     = (const float*)d_in[7];
  const float* W1        = (const float*)d_in[8];
  const float* b1        = (const float*)d_in[9];
  const float* W2        = (const float*)d_in[10];
  const float* b2        = (const float*)d_in[11];
  const float* W_out     = (const float*)d_in[12];
  const float* b_out     = (const float*)d_in[13];
  float*  out   = (float*)d_out;
  __bf16* Bfrag = (__bf16*)d_ws;   // 40*8*32*16 bf16 = 320 KB

  prep_wcomp_kernel<<<320, 512, 0, stream>>>(W_comp, Bfrag);

  size_t smem = (size_t)(65 * MODEL + 64 * MODEL + 4 * 64 + 2 * MLPD) * sizeof(float);
  pyramid_kernel<<<64, 256, smem, stream>>>(sentences, embed, W_enc, b_enc,
                                            b_comp, W_sel, b_sel,
                                            W1, b1, W2, b2, W_out, b_out,
                                            Bfrag, out);
}
